// GPTAttention_86861418594475
// MI455X (gfx1250) — matrix operand order repaired and checked by reading, not compiled, not run on hardware
//
#include <hip/hip_runtime.h>

typedef _Float16 half_t;
typedef __attribute__((ext_vector_type(16))) _Float16 v16h;
typedef __attribute__((ext_vector_type(8)))  _Float16 v8h;
typedef __attribute__((ext_vector_type(8)))  float    v8f;

#define BB 4
#define TT 2048
#define CC 1024
#define NHH 16
#define HDD 64
#define MM (BB*TT)      /* 8192 */
#define C3 (3*CC)       /* 3072 */

__device__ __forceinline__ v16h cat8(v8h lo, v8h hi) {
  return __builtin_shufflevector(lo, hi, 0,1,2,3,4,5,6,7,8,9,10,11,12,13,14,15);
}
__device__ __forceinline__ v8h ld8(const half_t* p) { return *(const v8h*)p; }

#define WMMA_F16(A,B,C) \
  __builtin_amdgcn_wmma_f32_16x16x32_f16(false, (A), false, (B), (short)0, (C), false, false)

// ---------------------------------------------------------------- fp32 -> fp16
__global__ void k_f2h(const float* __restrict__ s, half_t* __restrict__ d, int n) {
  int i = blockIdx.x * blockDim.x + threadIdx.x;
  int st = gridDim.x * blockDim.x;
  for (; i < n; i += st) d[i] = (half_t)s[i];
}

// ---------------------------------------------------------------- WMMA GEMM
// A: [M, CC] f16 row-major. W: [CC, NSTRIDE] f16 row-major.
// Block = 8 waves * (16 M x 64 N) = 128 x 64 output tile.
// Per K-step (32): block stages a 32x64 B slab into fragment-swizzled LDS
// (ping-pong double buffered, one barrier per step); each wave loads one A
// fragment and issues 4 WMMAs against the 4 column sub-slabs.
// MODE 0: qkv epilogue (split q/k/v, q pre-scaled, v transposed).
// MODE 1: f32 output (+bias) row-major.
template<int NSTRIDE, int MODE>
__global__ __launch_bounds__(256) void k_gemm(
    const half_t* __restrict__ ah, const half_t* __restrict__ wh,
    const float* __restrict__ bias,
    half_t* __restrict__ qh, half_t* __restrict__ kh, half_t* __restrict__ vth,
    float* __restrict__ outf)
{
  // [buf][c-subtile][lane][e] halves; one c-subtile = 32 lanes * 16 halves
  __shared__ __align__(16) half_t sB[2][4 * 32 * 16];
  const int tid  = threadIdx.x;
  const int wave = tid >> 5;
  const int lane = tid & 31;
  const int hi   = lane >> 4;
  const int ln   = lane & 15;
  const int n0   = blockIdx.y * 64;
  const int rowM = blockIdx.x * 128 + wave * 16;
  const half_t* arow = ah + (size_t)(rowM + ln) * CC;

  // cooperative fill: thread -> one K row (frow), one 8-col segment (fseg)
  const int frow = tid >> 3;             // 0..31
  const int fseg = tid & 7;              // 0..7 -> cols fseg*8 .. +8
  const int fe   = frow & 15;            // fragment element index
  const int fgrp = (frow >> 4) << 4;     // lane-group offset (K>=16 -> +16)
  const half_t* wrow = wh + (size_t)frow * NSTRIDE + n0 + fseg * 8;

  auto fill = [&](half_t* buf, int k0) {
    v8h w8 = ld8(wrow + (size_t)k0 * NSTRIDE);   // coalesced 16B per thread
#pragma unroll
    for (int j = 0; j < 8; ++j) {
      const int colt = fseg * 8 + j;             // 0..63 in slab
      const int c  = colt >> 4;
      const int nn = colt & 15;
      buf[(c * 32 + fgrp + nn) * 16 + fe] = w8[j];
    }
  };

  v8f acc[4] = {};
  fill(sB[0], 0);
  __syncthreads();
  for (int k0 = 0; k0 < CC; k0 += 32) {
    const int cur = (k0 >> 5) & 1;
    if (k0 + 32 < CC) {
      fill(sB[cur ^ 1], k0 + 32);                // overlap with WMMAs below
      if (k0 + 64 < CC)
        __builtin_prefetch(wrow + (size_t)(k0 + 64) * NSTRIDE, 0, 1);
    }
    // A fragment: lane row = ln; runs K = k0+hi*16 .. k0+hi*16+15
    v16h a = cat8(ld8(arow + k0 + hi * 16), ld8(arow + k0 + hi * 16 + 8));
    const half_t* bb = sB[cur];
#pragma unroll
    for (int c = 0; c < 4; ++c) {
      const half_t* bp = bb + (c * 32 + lane) * 16;
      v16h b = cat8(ld8(bp), ld8(bp + 8));
      acc[c] = WMMA_F16(a, b, acc[c]);
    }
    __syncthreads();
  }

#pragma unroll
  for (int c = 0; c < 4; ++c) {
    const int col = n0 + c * 16 + ln;
    const float bv = bias[col];
    if (MODE == 0) {
      const int region = col >> 10;        // 0=q 1=k 2=v
      const int cj = col & (CC - 1);
      const int h = cj >> 6;
      const int d = cj & 63;
#pragma unroll
      for (int r = 0; r < 8; ++r) {
        const int m  = rowM + r + 8 * hi;  // C layout: VGPR r -> row r+8*hi
        const int bI = m >> 11;
        const int t  = m & (TT - 1);
        const float val = acc[c][r] + bv;
        const size_t bh = (size_t)(bI * NHH + h);
        if (region == 0)      qh[(bh * TT + t) * HDD + d] = (half_t)(val * 0.125f);
        else if (region == 1) kh[(bh * TT + t) * HDD + d] = (half_t)val;
        else                  vth[(bh * HDD + d) * TT + t] = (half_t)val;
      }
    } else {
#pragma unroll
      for (int r = 0; r < 8; ++r) {
        const int m = rowM + r + 8 * hi;
        outf[(size_t)m * CC + col] = acc[c][r] + bv;
      }
    }
  }
}

// ---------------------------------------------------------------- flash attention
// One wave per (b,h, 16-row q tile). Online softmax over 32-key blocks.
__global__ __launch_bounds__(256) void k_attn(
    const half_t* __restrict__ qh, const half_t* __restrict__ kh,
    const half_t* __restrict__ vth, half_t* __restrict__ yh)
{
  __shared__ __align__(16) half_t sP[8 * 16 * 32];   // per-wave P re-swizzle patch
  const int wave = threadIdx.x >> 5;
  const int lane = threadIdx.x & 31;
  const int hi = lane >> 4, ln = lane & 15;
  const int task = blockIdx.x * 8 + wave;
  const int bh = task >> 7;                 // b*NH + h
  const int qt = task & 127;                // q tile index
  half_t* pP = sP + wave * (16 * 32);

  // Q A-fragments (row = ln, dims 0..31 / 32..63)
  const half_t* qrow = qh + ((size_t)bh * TT + qt * 16 + ln) * HDD;
  const v16h aq0 = cat8(ld8(qrow + hi * 16),      ld8(qrow + hi * 16 + 8));
  const v16h aq1 = cat8(ld8(qrow + 32 + hi * 16), ld8(qrow + 32 + hi * 16 + 8));

  v8f o0 = {}, o1 = {}, o2 = {}, o3 = {};
  float mrow[8], lrow[8];
#pragma unroll
  for (int r = 0; r < 8; ++r) { mrow[r] = -1e30f; lrow[r] = 0.f; }

  const int limit = (qt + 1) * 16;
  const int qr0 = qt * 16 + 8 * hi;
  for (int kb = 0; kb < limit; kb += 32) {
    // S = Q @ K^T for 32 keys (two 16-key C tiles)
    v8f s0 = {}, s1 = {};
    {
      const half_t* k0r = kh + ((size_t)bh * TT + kb + ln) * HDD;     // key = kb+ln
      v16h b0 = cat8(ld8(k0r + hi * 16),      ld8(k0r + hi * 16 + 8));       // dims 0..31
      v16h b1 = cat8(ld8(k0r + 32 + hi * 16), ld8(k0r + 32 + hi * 16 + 8));  // dims 32..63
      s0 = WMMA_F16(aq0, b0, s0);
      s0 = WMMA_F16(aq1, b1, s0);
      const half_t* k1r = k0r + 16 * HDD;                              // key = kb+16+ln
      v16h c0 = cat8(ld8(k1r + hi * 16),      ld8(k1r + hi * 16 + 8));
      v16h c1 = cat8(ld8(k1r + 32 + hi * 16), ld8(k1r + 32 + hi * 16 + 8));
      s1 = WMMA_F16(aq0, c0, s1);
      s1 = WMMA_F16(aq1, c1, s1);
    }
    // causal mask + online softmax (rows per-VGPR, keys per-lane)
    const int key0 = kb + ln, key1 = key0 + 16;
    float p0[8], p1[8];
#pragma unroll
    for (int r = 0; r < 8; ++r) {
      const int qr = qr0 + r;
      float v0 = (key0 <= qr) ? s0[r] : -1e30f;
      float v1 = (key1 <= qr) ? s1[r] : -1e30f;
      float mx = fmaxf(v0, v1);
      mx = fmaxf(mx, __shfl_xor(mx, 1, 32));
      mx = fmaxf(mx, __shfl_xor(mx, 2, 32));
      mx = fmaxf(mx, __shfl_xor(mx, 4, 32));
      mx = fmaxf(mx, __shfl_xor(mx, 8, 32));
      const float mn = fmaxf(mrow[r], mx);
      const float sc = __expf(mrow[r] - mn);
      mrow[r] = mn;
      p0[r] = __expf(v0 - mn);
      p1[r] = __expf(v1 - mn);
      float sm = p0[r] + p1[r];
      sm += __shfl_xor(sm, 1, 32);
      sm += __shfl_xor(sm, 2, 32);
      sm += __shfl_xor(sm, 4, 32);
      sm += __shfl_xor(sm, 8, 32);
      lrow[r] = lrow[r] * sc + sm;
      o0[r] *= sc; o1[r] *= sc; o2[r] *= sc; o3[r] *= sc;
      // spill P tile (C layout) to LDS as [row][key]
      pP[(8 * hi + r) * 32 + ln]      = (half_t)p0[r];
      pP[(8 * hi + r) * 32 + 16 + ln] = (half_t)p1[r];
    }
    asm volatile("s_wait_dscnt 0" ::: "memory");   // wave-local LDS RAW fence
    // reload P as 16x32 A fragment
    const half_t* prow = pP + ln * 32;
    const v16h ap = cat8(ld8(prow + hi * 16), ld8(prow + hi * 16 + 8));
    // O += P @ V  (V stored transposed [bh][d][t] -> contiguous B fragments)
    const half_t* vb = vth + (size_t)bh * HDD * TT + kb + hi * 16;
    {
      v16h bv0 = cat8(ld8(vb + (size_t)(ln)      * TT), ld8(vb + (size_t)(ln)      * TT + 8));
      o0 = WMMA_F16(ap, bv0, o0);
      v16h bv1 = cat8(ld8(vb + (size_t)(16 + ln) * TT), ld8(vb + (size_t)(16 + ln) * TT + 8));
      o1 = WMMA_F16(ap, bv1, o1);
      v16h bv2 = cat8(ld8(vb + (size_t)(32 + ln) * TT), ld8(vb + (size_t)(32 + ln) * TT + 8));
      o2 = WMMA_F16(ap, bv2, o2);
      v16h bv3 = cat8(ld8(vb + (size_t)(48 + ln) * TT), ld8(vb + (size_t)(48 + ln) * TT + 8));
      o3 = WMMA_F16(ap, bv3, o3);
    }
  }
  // finalize: O / l, write back as [b][t][C] f16 for the projection GEMM
  const int b = bh >> 4, h = bh & 15;
#pragma unroll
  for (int r = 0; r < 8; ++r) {
    const float inv = 1.0f / lrow[r];
    const int t = qt * 16 + 8 * hi + r;
    half_t* yb = yh + ((size_t)b * TT + t) * CC + h * HDD;
    yb[ln]      = (half_t)(o0[r] * inv);
    yb[16 + ln] = (half_t)(o1[r] * inv);
    yb[32 + ln] = (half_t)(o2[r] * inv);
    yb[48 + ln] = (half_t)(o3[r] * inv);
  }
}

// ---------------------------------------------------------------- launch
extern "C" void kernel_launch(void* const* d_in, const int* in_sizes, int n_in,
                              void* d_out, int out_size, void* d_ws, size_t ws_size,
                              hipStream_t stream) {
  (void)in_sizes; (void)n_in; (void)out_size; (void)ws_size;
  const float* x      = (const float*)d_in[0];
  const float* w_attn = (const float*)d_in[1];
  const float* b_attn = (const float*)d_in[2];
  const float* w_proj = (const float*)d_in[3];
  const float* b_proj = (const float*)d_in[4];
  float* out = (float*)d_out;

  char* p = (char*)d_ws;
  half_t* xh  = (half_t*)p; p += (size_t)MM * CC * 2;   // 16 MB
  half_t* wah = (half_t*)p; p += (size_t)CC * C3 * 2;   //  6 MB
  half_t* wph = (half_t*)p; p += (size_t)CC * CC * 2;   //  2 MB
  half_t* qh  = (half_t*)p; p += (size_t)MM * CC * 2;   // 16 MB  [bh][t][d], pre-scaled
  half_t* kh  = (half_t*)p; p += (size_t)MM * CC * 2;   // 16 MB  [bh][t][d]
  half_t* vth = (half_t*)p; p += (size_t)MM * CC * 2;   // 16 MB  [bh][d][t]
  half_t* yh  = (half_t*)p; p += (size_t)MM * CC * 2;   // 16 MB  [b][t][C]

  k_f2h<<<2048, 256, 0, stream>>>(x, xh, MM * CC);
  k_f2h<<<2048, 256, 0, stream>>>(w_attn, wah, CC * C3);
  k_f2h<<<1024, 256, 0, stream>>>(w_proj, wph, CC * CC);

  k_gemm<C3, 0><<<dim3(MM / 128, C3 / 64), 256, 0, stream>>>(
      xh, wah, b_attn, qh, kh, vth, nullptr);

  k_attn<<<(BB * NHH * (TT / 16)) / 8, 256, 0, stream>>>(qh, kh, vth, yh);

  k_gemm<CC, 1><<<dim3(MM / 128, CC / 64), 256, 0, stream>>>(
      yh, wph, b_proj, nullptr, nullptr, nullptr, out);
}
